// TimeAwareIMULSTMEncoder_14259291423004
// MI455X (gfx1250) — matrix operand order repaired and checked by reading, not compiled, not run on hardware
//
#include <hip/hip_runtime.h>
#include <hip/hip_bf16.h>

// ---------- bf16 helpers (round-to-nearest-even) ----------
typedef __attribute__((ext_vector_type(16))) __bf16 v16bf;
typedef __attribute__((ext_vector_type(8)))  float  v8f;

__device__ __forceinline__ unsigned short f2bf(float f) {
    unsigned int u = __float_as_uint(f);
    u += 0x7FFFu + ((u >> 16) & 1u);
    return (unsigned short)(u >> 16);
}
__device__ __forceinline__ float bf2f(unsigned short h) {
    return __uint_as_float(((unsigned int)h) << 16);
}
// one v_exp + one v_rcp each
__device__ __forceinline__ float fast_sigmoid(float x) {
    return __frcp_rn(1.0f + __expf(-x));
}
__device__ __forceinline__ float fast_tanh(float x) {
    return fmaf(2.0f, __frcp_rn(1.0f + __expf(-2.0f * x)), -1.0f);
}

union AFrag { v16bf v; uint4 q[2]; unsigned short u[16]; };
union CFrag { v8f  v; float f[8]; float4 q[2]; };

// ---------- f32 [K][N] -> bf16 transposed [N][K] ----------
__global__ void f2bf_t_kernel(const float* __restrict__ in, unsigned short* __restrict__ out,
                              int Nn, int Kk)
{
    int i = blockIdx.x * 256 + threadIdx.x;
    if (i < Nn * Kk) {
        int n = i / Kk, k = i - n * Kk;
        out[i] = f2bf(in[(size_t)k * Nn + n]);
    }
}

// ---------- input projection: [imu(6), t, rate] @ W_in + b_in, relu -> bf16 [64000][64] ----------
__global__ __launch_bounds__(256)
void input_proj(const float* __restrict__ imu, const float* __restrict__ Win,
                const float* __restrict__ bin, unsigned short* __restrict__ X)
{
    int gid = blockIdx.x * 256 + threadIdx.x;      // 64000*64 threads
    if (gid >= 64000 * 64) return;
    int row = gid >> 6;
    int cc  = gid & 63;
    int tt  = row % 50;
    float x[8];
#pragma unroll
    for (int k = 0; k < 6; ++k) x[k] = imu[(size_t)row * 6 + k];
    x[6] = (float)tt * (1.0f / 49.0f);             // linspace(0,1,50)
    x[7] = 1.0f;                                   // L / EXPECTED = 50/50
    float s = bin[cc];
#pragma unroll
    for (int k = 0; k < 8; ++k) s += x[k] * Win[k * 64 + cc];
    X[gid] = f2bf(fmaxf(s, 0.0f));
}

// ---------- tiled WMMA GEMM: C[M,N] = A[M,K](bf16) @ B^T[N,K](bf16) + bias0 + bias1 ----------
// block = 256 threads (8 waves: 4 M-strips x 2 N-halves), WG tile 64(M) x 128(N).
// OMODE: 0 = f32 row-major, 2 = bf16 "lstm fragment order" gates buffer.
template<bool RELU, int OMODE>
__global__ __launch_bounds__(256)
void gemm_bf16(const unsigned short* __restrict__ A, int lda,
               const unsigned short* __restrict__ BT,              // [N][K]
               const float* __restrict__ bias0, const float* __restrict__ bias1,
               float* __restrict__ Cf, unsigned short* __restrict__ Cb, int ldc,
               int K)
{
    __shared__ __align__(16) unsigned short As[64 * 32];   // [m][k]
    __shared__ __align__(16) unsigned short Bs[128 * 32];  // [n][k]  (transposed tile)
    const int tid   = threadIdx.x;
    const int lane  = tid & 31;
    const int wave  = tid >> 5;
    const int wm    = wave >> 1;               // 0..3  (M strip)
    const int wn    = wave & 1;                // 0..1  (N half)
    const int mBase = blockIdx.y * 64;
    const int nBase = blockIdx.x * 128;
    const int lhi   = lane >> 4;               // 0/1
    const int llo   = lane & 15;

    CFrag acc[4];
#pragma unroll
    for (int j = 0; j < 4; ++j)
#pragma unroll
        for (int v = 0; v < 8; ++v) acc[j].f[v] = 0.0f;

    for (int k0 = 0; k0 < K; k0 += 32) {
        {   // A tile: 64x32 = 2048 ushorts, one uint4 per thread
            int flat = tid * 8;
            int m = flat >> 5, k = flat & 31;
            *(uint4*)&As[flat] = *(const uint4*)&A[(size_t)(mBase + m) * lda + k0 + k];
        }
        {   // B tile (transposed): 128x32 = 4096 ushorts, two uint4 per thread
            int flat = tid * 16;
            int n = flat >> 5, koff = flat & 31;
            const unsigned short* src = &BT[(size_t)(nBase + n) * K + k0 + koff];
            *(uint4*)&Bs[n * 32 + koff]     = *(const uint4*)&src[0];
            *(uint4*)&Bs[n * 32 + koff + 8] = *(const uint4*)&src[8];
        }
        __syncthreads();

        AFrag a;   // ISA 16-bit A 16x32 layout: two contiguous 8-half runs per lane
        {
            const unsigned short* ap = &As[(wm * 16 + llo) * 32 + (lhi << 3)];
            a.q[0] = *(const uint4*)&ap[0];
            a.q[1] = *(const uint4*)&ap[16];
        }
#pragma unroll
        for (int j = 0; j < 4; ++j) {
            AFrag b;   // ISA 16-bit B 32x16 layout: two contiguous 8-half K-runs per lane (same as A)
            const unsigned short* bp = &Bs[(wn * 64 + j * 16 + llo) * 32 + (lhi << 3)];
            b.q[0] = *(const uint4*)&bp[0];
            b.q[1] = *(const uint4*)&bp[16];
            acc[j].v = __builtin_amdgcn_wmma_f32_16x16x32_bf16(
                false, a.v, false, b.v, (short)0, acc[j].v, false, false);
        }
        __syncthreads();
    }

#pragma unroll
    for (int j = 0; j < 4; ++j)
#pragma unroll
        for (int v = 0; v < 8; ++v) {
            int m = wm * 16 + (lhi << 3) + v;
            int n = wn * 64 + j * 16 + llo;
            int gm = mBase + m, gn = nBase + n;
            float val = acc[j].f[v];
            if (bias0) val += bias0[gn];
            if (bias1) val += bias1[gn];
            if (RELU) val = fmaxf(val, 0.0f);
            if constexpr (OMODE == 0) {
                Cf[(size_t)gm * ldc + gn] = val;
            } else {
                // scatter into lstm fragment order: [nbTile][te][ (gn>>4)*256 + lane'*8 + v' ]
                int nb = gm / 50, te = gm - nb * 50;
                size_t idx = ((size_t)(nb >> 4) * 50 + te) * 8192
                           + (size_t)(gn >> 4) * 256
                           + ((nb & 15) >> 3) * 128
                           + (gn & 15) * 8
                           + (nb & 7);
                Cb[idx] = f2bf(val);
            }
        }
}

// ---------- recurrent BiLSTM layer ----------
// grid = (80 batch tiles of 16, 2 directions); block = 256 (8 waves).
// Whh^T fragments live in REGISTERS for all 50 steps (16 frags = 128 VGPRs/lane).
// LDS: h tile (16x128 bf16 = 4KB) + gate frag buffer (8192 f32 = 32KB).
__global__ __launch_bounds__(256)
void lstm_rec(const unsigned short* __restrict__ gatesF, // [2][80][50][8192] bf16, fragment order
              const unsigned short* __restrict__ WhhT,   // [2][512][128] bf16 (transposed)
              unsigned short* __restrict__ Y)            // [64000][256] bf16 row-major
{
    __shared__ __align__(16) unsigned short hA[16 * 128];   // [r][k]
    __shared__ __align__(16) float gl[8192];                // fragment-order gate staging

    const int tid  = threadIdx.x;
    const int lane = tid & 31;
    const int wn   = tid >> 5;                 // wave owns gate cols [wn*64, wn*64+64)
    const int lhi  = lane >> 4;
    const int llo  = lane & 15;
    const int dir  = blockIdx.y;
    const unsigned short* gBase =
        gatesF + ((size_t)dir * 80 + blockIdx.x) * 50 * 8192;
    const unsigned short* W = WhhT + (size_t)dir * 512 * 128;

    // load all 16 Whh B-fragments once, straight into registers
    AFrag Bf[4][4];   // [kk][j]
#pragma unroll
    for (int kk = 0; kk < 4; ++kk)
#pragma unroll
        for (int j = 0; j < 4; ++j) {
            const unsigned short* bp =
                &W[(wn * 64 + j * 16 + llo) * 128 + kk * 32 + (lhi << 3)];
            Bf[kk][j].q[0] = *(const uint4*)&bp[0];
            Bf[kk][j].q[1] = *(const uint4*)&bp[16];
        }

    {   // zero h tile
        uint4 z; z.x = z.y = z.z = z.w = 0u;
        *(uint4*)&hA[tid * 8] = z;
    }
    // elementwise ownership: thread owns 8 CONSECUTIVE hidden units of one row
    const int er  = (tid * 8) >> 7;            // row 0..15
    const int eu0 = (tid * 8) & 127;           // first hidden unit (multiple of 8)
    float c[8];
#pragma unroll
    for (int i = 0; i < 8; ++i) c[i] = 0.0f;
    __syncthreads();

    for (int t = 0; t < 50; ++t) {
        const int te = dir ? (49 - t) : t;     // bwd direction walks time reversed

        // issue precomputed input-side gate loads early (one b128 per fragment)
        uint4 gv[4];
#pragma unroll
        for (int j = 0; j < 4; ++j)
            gv[j] = *(const uint4*)&gBase[(size_t)te * 8192 + ((wn * 4 + j) * 32 + lane) * 8];

        // hoist all 4 A fragments (h tile) before the WMMA chain
        AFrag a[4];
#pragma unroll
        for (int kk = 0; kk < 4; ++kk) {
            const unsigned short* ap = &hA[llo * 128 + kk * 32 + (lhi << 3)];
            a[kk].q[0] = *(const uint4*)&ap[0];
            a[kk].q[1] = *(const uint4*)&ap[16];
        }

        // gates_h = h @ Whh : 16 back-to-back WMMAs, B operands register-resident
        CFrag acc[4];
#pragma unroll
        for (int j = 0; j < 4; ++j)
#pragma unroll
            for (int v = 0; v < 8; ++v) acc[j].f[v] = 0.0f;
#pragma unroll
        for (int kk = 0; kk < 4; ++kk)
#pragma unroll
            for (int j = 0; j < 4; ++j)
                acc[j].v = __builtin_amdgcn_wmma_f32_16x16x32_bf16(
                    false, a[kk].v, false, Bf[kk][j].v, (short)0, acc[j].v, false, false);

        // add input-side gates, spill to LDS in fragment order (2x b128 per fragment)
#pragma unroll
        for (int j = 0; j < 4; ++j) {
            unsigned int ww[4] = {gv[j].x, gv[j].y, gv[j].z, gv[j].w};
            CFrag r;
#pragma unroll
            for (int v = 0; v < 8; ++v) {
                unsigned short gb = (unsigned short)((ww[v >> 1] >> ((v & 1) * 16)) & 0xffffu);
                r.f[v] = acc[j].f[v] + bf2f(gb);
            }
            float* dst = &gl[((wn * 4 + j) * 32 + lane) * 8];
            *(float4*)&dst[0] = r.q[0];
            *(float4*)&dst[4] = r.q[1];
        }
        __syncthreads();

        // cell update (i,f,g,o): gl index for gate col n, row r:
        //   (n>>4)*256 + ((r>>3)*16 + (n&15))*8 + (r&7)
        const int rbit = ((er >> 3) * 16) * 8 + (er & 7);
        unsigned short hb[8];
#pragma unroll
        for (int i = 0; i < 8; ++i) {
            int u = eu0 + i;
#define GL_AT(nn) gl[(((nn) >> 4) * 256) + rbit + ((nn) & 15) * 8]
            float ig = fast_sigmoid(GL_AT(u));
            float fg = fast_sigmoid(GL_AT(u + 128));
            float gg = fast_tanh   (GL_AT(u + 256));
            float og = fast_sigmoid(GL_AT(u + 384));
#undef GL_AT
            c[i] = fg * c[i] + ig * gg;
            hb[i] = f2bf(og * fast_tanh(c[i]));
        }
        // vectorized write-back: 8 consecutive bf16 -> one b128 to LDS and to global
        uint4 hq;
        hq.x = (unsigned)hb[0] | ((unsigned)hb[1] << 16);
        hq.y = (unsigned)hb[2] | ((unsigned)hb[3] << 16);
        hq.z = (unsigned)hb[4] | ((unsigned)hb[5] << 16);
        hq.w = (unsigned)hb[6] | ((unsigned)hb[7] << 16);
        *(uint4*)&hA[er * 128 + eu0] = hq;
        *(uint4*)&Y[((size_t)(blockIdx.x * 16 + er) * 50 + te) * 256 + dir * 128 + eu0] = hq;
        __syncthreads();
    }
}

// ---------- aggregation: mean / max over t + final hidden -> agg bf16 [1280][768] ----------
__global__ __launch_bounds__(256)
void aggregate(const unsigned short* __restrict__ Y, unsigned short* __restrict__ agg)
{
    int n  = blockIdx.x;
    int cc = threadIdx.x;                      // 0..255
    float s = 0.0f, mx = -3.4e38f;
    for (int t = 0; t < 50; ++t) {
        float v = bf2f(Y[((size_t)n * 50 + t) * 256 + cc]);
        s += v; mx = fmaxf(mx, v);
    }
    agg[(size_t)n * 768 + cc]       = f2bf(s * (1.0f / 50.0f));
    agg[(size_t)n * 768 + 256 + cc] = f2bf(mx);
    unsigned short fv = (cc < 128) ? Y[((size_t)n * 50 + 49) * 256 + cc]   // fwd h at t=L-1
                                   : Y[((size_t)n * 50 +  0) * 256 + cc];  // bwd h at t=0
    agg[(size_t)n * 768 + 512 + cc] = fv;
}

// ---------- LayerNorm(512) + ReLU -> bf16 ----------
__global__ __launch_bounds__(256)
void ln_relu(const float* __restrict__ Hf, const float* __restrict__ g,
             const float* __restrict__ b, unsigned short* __restrict__ O)
{
    __shared__ float red[256];
    int n = blockIdx.x, tid = threadIdx.x;
    float v0 = Hf[(size_t)n * 512 + tid];
    float v1 = Hf[(size_t)n * 512 + 256 + tid];
    red[tid] = v0 + v1;
    __syncthreads();
    for (int s = 128; s > 0; s >>= 1) { if (tid < s) red[tid] += red[tid + s]; __syncthreads(); }
    float mu = red[0] * (1.0f / 512.0f);
    __syncthreads();
    float d0 = v0 - mu, d1 = v1 - mu;
    red[tid] = d0 * d0 + d1 * d1;
    __syncthreads();
    for (int s = 128; s > 0; s >>= 1) { if (tid < s) red[tid] += red[tid + s]; __syncthreads(); }
    float rstd = rsqrtf(red[0] * (1.0f / 512.0f) + 1e-5f);
    O[(size_t)n * 512 + tid]       = f2bf(fmaxf(d0 * rstd * g[tid]       + b[tid],       0.0f));
    O[(size_t)n * 512 + 256 + tid] = f2bf(fmaxf(d1 * rstd * g[tid + 256] + b[tid + 256], 0.0f));
}

// =====================================================================
extern "C" void kernel_launch(void* const* d_in, const int* in_sizes, int n_in,
                              void* d_out, int out_size, void* d_ws, size_t ws_size,
                              hipStream_t stream)
{
    (void)in_sizes; (void)n_in; (void)out_size; (void)ws_size;

    const float* imu   = (const float*)d_in[0];
    const float* W_in  = (const float*)d_in[1];
    const float* b_in  = (const float*)d_in[2];
    const float* Wih0  = (const float*)d_in[3];
    const float* Whh0  = (const float*)d_in[4];
    const float* bih0  = (const float*)d_in[5];
    const float* bhh0  = (const float*)d_in[6];
    const float* Wih1  = (const float*)d_in[7];
    const float* Whh1  = (const float*)d_in[8];
    const float* bih1  = (const float*)d_in[9];
    const float* bhh1  = (const float*)d_in[10];
    const float* Wih2  = (const float*)d_in[11];
    const float* Whh2  = (const float*)d_in[12];
    const float* bih2  = (const float*)d_in[13];
    const float* bhh2  = (const float*)d_in[14];
    const float* Wout1 = (const float*)d_in[15];
    const float* bout1 = (const float*)d_in[16];
    const float* ln_g  = (const float*)d_in[17];
    const float* ln_b  = (const float*)d_in[18];
    const float* Wout2 = (const float*)d_in[19];
    const float* bout2 = (const float*)d_in[20];
    float* out = (float*)d_out;

    // ---- workspace carve-up (256B aligned) ----
    char* ws = (char*)d_ws;
    size_t off = 0;
    auto take = [&](size_t bytes) -> char* {
        char* p = ws + off;
        off = (off + bytes + 255) & ~(size_t)255;
        return p;
    };
    unsigned short* xin0   = (unsigned short*)take((size_t)64000 * 64 * 2);
    unsigned short* YA     = (unsigned short*)take((size_t)64000 * 256 * 2);
    unsigned short* YB     = (unsigned short*)take((size_t)64000 * 256 * 2);
    unsigned short* gatesF = (unsigned short*)take((size_t)2 * 80 * 50 * 8192 * 2);
    unsigned short* aggB   = (unsigned short*)take((size_t)1280 * 768 * 2);
    float*          h1f    = (float*)         take((size_t)1280 * 512 * 4);
    unsigned short* h1nB   = (unsigned short*)take((size_t)1280 * 512 * 2);
    unsigned short* WihT0  = (unsigned short*)take((size_t)2 * 512 * 64  * 2);  // [N][K]
    unsigned short* WhhT0  = (unsigned short*)take((size_t)2 * 512 * 128 * 2);
    unsigned short* WihT1  = (unsigned short*)take((size_t)2 * 512 * 256 * 2);
    unsigned short* WhhT1  = (unsigned short*)take((size_t)2 * 512 * 128 * 2);
    unsigned short* WihT2  = (unsigned short*)take((size_t)2 * 512 * 256 * 2);
    unsigned short* WhhT2  = (unsigned short*)take((size_t)2 * 512 * 128 * 2);
    unsigned short* Wout1T = (unsigned short*)take((size_t)512 * 768 * 2);
    unsigned short* Wout2T = (unsigned short*)take((size_t)256 * 512 * 2);

    // transpose-convert weights: in [K][N] f32 -> out [N][K] bf16 (per direction slice)
    auto convT = [&](const float* src, unsigned short* dst, int Nn, int Kk) {
        int n = Nn * Kk;
        f2bf_t_kernel<<<(n + 255) / 256, 256, 0, stream>>>(src, dst, Nn, Kk);
    };
    for (int d = 0; d < 2; ++d) {
        convT(Wih0 + (size_t)d * 64  * 512, WihT0 + (size_t)d * 512 * 64,  512, 64);
        convT(Whh0 + (size_t)d * 128 * 512, WhhT0 + (size_t)d * 512 * 128, 512, 128);
        convT(Wih1 + (size_t)d * 256 * 512, WihT1 + (size_t)d * 512 * 256, 512, 256);
        convT(Whh1 + (size_t)d * 128 * 512, WhhT1 + (size_t)d * 512 * 128, 512, 128);
        convT(Wih2 + (size_t)d * 256 * 512, WihT2 + (size_t)d * 512 * 256, 512, 256);
        convT(Whh2 + (size_t)d * 128 * 512, WhhT2 + (size_t)d * 512 * 128, 512, 128);
    }
    convT(Wout1, Wout1T, 512, 768);
    convT(Wout2, Wout2T, 256, 512);

    // input projection -> bf16 [64000][64]
    input_proj<<<(64000 * 64) / 256, 256, 0, stream>>>(imu, W_in, b_in, xin0);

    const dim3 gGate(4, 1000);   // N=512/128, M=64000/64
    const dim3 gRec(80, 2);      // 80 batch tiles x 2 directions
    const size_t gateStride = (size_t)80 * 50 * 8192;

    // ---- layer 0 ----
    for (int d = 0; d < 2; ++d)
        gemm_bf16<false, 2><<<gGate, 256, 0, stream>>>(
            xin0, 64, WihT0 + (size_t)d * 512 * 64,
            bih0 + d * 512, bhh0 + d * 512,
            nullptr, gatesF + d * gateStride, 0, 64);
    lstm_rec<<<gRec, 256, 0, stream>>>(gatesF, WhhT0, YA);

    // ---- layer 1 ----
    for (int d = 0; d < 2; ++d)
        gemm_bf16<false, 2><<<gGate, 256, 0, stream>>>(
            YA, 256, WihT1 + (size_t)d * 512 * 256,
            bih1 + d * 512, bhh1 + d * 512,
            nullptr, gatesF + d * gateStride, 0, 256);
    lstm_rec<<<gRec, 256, 0, stream>>>(gatesF, WhhT1, YB);

    // ---- layer 2 ----
    for (int d = 0; d < 2; ++d)
        gemm_bf16<false, 2><<<gGate, 256, 0, stream>>>(
            YB, 256, WihT2 + (size_t)d * 512 * 256,
            bih2 + d * 512, bhh2 + d * 512,
            nullptr, gatesF + d * gateStride, 0, 256);
    lstm_rec<<<gRec, 256, 0, stream>>>(gatesF, WhhT2, YA);

    // ---- aggregation + head ----
    aggregate<<<1280, 256, 0, stream>>>(YA, aggB);

    gemm_bf16<false, 0><<<dim3(4, 20), 256, 0, stream>>>(
        aggB, 768, Wout1T, bout1, nullptr, h1f, nullptr, 512, 768);

    ln_relu<<<1280, 256, 0, stream>>>(h1f, ln_g, ln_b, h1nB);

    gemm_bf16<false, 0><<<dim3(2, 20), 256, 0, stream>>>(
        h1nB, 512, Wout2T, bout2, nullptr, out, nullptr, 256, 512);
}
